// Synonym_41386304864593
// MI455X (gfx1250) — matrix operand order repaired and checked by reading, not compiled, not run on hardware
//
#include <hip/hip_runtime.h>

// ---------------- problem constants (match reference) ----------------
#define VOCAB   400000
#define DIM     300
#define NQ      512
#define TOPK    5
#define KP      320            // DIM padded to multiple of 32 (WMMA K); tail zero-filled
#define KSTEPS  (KP / 32)      // 10
#define VCHUNK  256            // vocab rows per chunk (per workgroup)
#define NTILES  (VCHUNK / 16)  // 16 N-tiles per chunk
#define CHUNKS  ((VOCAB + VCHUNK - 1) / VCHUNK)  // 1563
#define QBLOCKS 4              // 4 * (8 waves * 16 q) = 512 queries

typedef __attribute__((ext_vector_type(16))) __bf16 v16bf;
typedef __attribute__((ext_vector_type(8)))  __bf16 v8bf;
typedef __attribute__((ext_vector_type(2)))  __bf16 v2bf;
typedef __attribute__((ext_vector_type(8)))  float  v8f;

// Register-resident top-5 insert with only static indices (no scratch spills).
__device__ __forceinline__ void insert5(float (&tv)[5], int (&ti)[5], float s, int v) {
  if (s <= tv[4]) return;
  if (s > tv[3]) {
    tv[4] = tv[3]; ti[4] = ti[3];
    if (s > tv[2]) {
      tv[3] = tv[2]; ti[3] = ti[2];
      if (s > tv[1]) {
        tv[2] = tv[1]; ti[2] = ti[1];
        if (s > tv[0]) { tv[1] = tv[0]; ti[1] = ti[0]; tv[0] = s; ti[0] = v; }
        else          { tv[1] = s;     ti[1] = v; }
      } else          { tv[2] = s;     ti[2] = v; }
    } else            { tv[3] = s;     ti[3] = v; }
  } else              { tv[4] = s;     ti[4] = v; }
}

// ---------------- K1: fused row-normalize + f32->bf16 convert ----------------
// One wave per row. Lane L owns element pairs k = 2L + 64*i (i=0..4) -> covers 0..319.
// Writes zero-padded KP-wide bf16 rows; packed pair stores (v_cvt_pk_bf16_f32 + b32 store).
__global__ __launch_bounds__(256) void norm_convert_kernel(const float* __restrict__ emb,
                                                           __bf16* __restrict__ ebf) {
  const int lane = threadIdx.x & 31;
  const int wave = threadIdx.x >> 5;
  const int row  = blockIdx.x * 8 + wave;
  if (row >= VOCAB) return;
  const float* p = emb + (size_t)row * DIM;

  float a[5], b[5];
  float s = 0.f;
  #pragma unroll
  for (int i = 0; i < 5; ++i) {
    int k = 2 * lane + 64 * i;
    a[i] = (k     < DIM) ? p[k]     : 0.f;
    b[i] = (k + 1 < DIM) ? p[k + 1] : 0.f;
    s += a[i] * a[i] + b[i] * b[i];
  }
  #pragma unroll
  for (int off = 16; off > 0; off >>= 1) s += __shfl_xor(s, off, 32);
  const float rn = rsqrtf(s);   // same in every lane after full butterfly

  __bf16* orow = ebf + (size_t)row * KP;
  #pragma unroll
  for (int i = 0; i < 5; ++i) {
    int k = 2 * lane + 64 * i;
    v2bf pk;
    pk.x = (__bf16)(a[i] * rn);
    pk.y = (__bf16)(b[i] * rn);
    *(v2bf*)(orow + k) = pk;
  }
}

// ---------------- K2: gather query rows out of the normalized bf16 table ----------------
__global__ __launch_bounds__(64) void gather_queries_kernel(const __bf16* __restrict__ ebf,
                                                            const int* __restrict__ ids,
                                                            __bf16* __restrict__ qbf) {
  const int q   = blockIdx.x;
  const int row = ids[q];
  const unsigned* src = (const unsigned*)(ebf + (size_t)row * KP);
  unsigned*       dst = (unsigned*)(qbf + (size_t)q * KP);
  for (int e = threadIdx.x; e < KP / 2; e += 64) dst[e] = src[e];   // bf16 pairs
}

// ---------------- K3: bf16 WMMA GEMM + per-chunk top-5 (pure matrix pipe) ----------------
// grid.x = QBLOCKS (query blocks of 128), grid.y = CHUNKS (vocab chunks of 256 rows)
// block  = 256 threads = 8 waves; each wave owns a 16-query M-tile.
// Per N-tile: all 20 B-fragment b128 loads are issued as one group, then a
// sched_barrier pins them above the 10 WMMAs -> loads drain with staggered
// partial loadcnt waits while the XDL pipe runs. No block barriers: each wave
// owns its LDS slice, so waves slip freely.
__global__ __launch_bounds__(256) void gemm_topk_kernel(const __bf16* __restrict__ ebf,
                                                        const __bf16* __restrict__ qbf,
                                                        float* __restrict__ pvals,
                                                        int*   __restrict__ pidx) {
  __shared__ float sh[8 * 256];   // per-wave 16x16 f32 accumulator staging

  const int lane  = threadIdx.x & 31;
  const int wave  = threadIdx.x >> 5;
  const int m     = lane & 15;    // M row within tile (A) / N col within tile (B)
  const int ksel  = lane >> 4;    // which K half this lane holds
  const int qbase = blockIdx.x * 128 + wave * 16;
  const int chunk = blockIdx.y;
  const int vbase = chunk * VCHUNK;

  // ---- A fragments (16x32 bf16 each, ISA layout), kept in registers all chunk ----
  // lane<16:  K = k0+0..7 and k0+16..23 ; lane>=16: K = k0+8..15 and k0+24..31
  v16bf afr[KSTEPS];
  const __bf16* qrow = qbf + (size_t)(qbase + m) * KP;
  #pragma unroll
  for (int t = 0; t < KSTEPS; ++t) {
    v8bf lo = *(const v8bf*)(qrow + 32 * t + ksel * 8);
    v8bf hi = *(const v8bf*)(qrow + 32 * t + 16 + ksel * 8);
    afr[t] = __builtin_shufflevector(lo, hi, 0,1,2,3,4,5,6,7,8,9,10,11,12,13,14,15);
  }

  float tv[5]; int ti[5];
  #pragma unroll
  for (int j = 0; j < 5; ++j) { tv[j] = -3.4e38f; ti[j] = 0; }

  float* shw = sh + wave * 256;

  for (int nt = 0; nt < NTILES; ++nt) {
    const int rowbase = vbase + nt * 16;
    int row = rowbase + m;
    if (row >= VOCAB) row = VOCAB - 1;            // clamp; masked out in top-k below
    const __bf16* brow = ebf + (size_t)row * KP + ksel * 8;

    // prefetch next tile's row segment into near cache (global_prefetch_b8)
    if (nt + 1 < NTILES) {
      int nrow = rowbase + 16 + m; if (nrow >= VOCAB) nrow = VOCAB - 1;
      __builtin_prefetch(ebf + (size_t)nrow * KP + ksel * 16, 0, 3);
    }

    // ---- issue ALL B-fragment loads for this tile (20x global_load_b128) ----
    v16bf bfr[KSTEPS];
    #pragma unroll
    for (int t = 0; t < KSTEPS; ++t) {
      v8bf lo = *(const v8bf*)(brow + 32 * t);
      v8bf hi = *(const v8bf*)(brow + 32 * t + 16);
      bfr[t] = __builtin_shufflevector(lo, hi, 0,1,2,3,4,5,6,7,8,9,10,11,12,13,14,15);
    }
    // Pin the load group above the WMMA group: forces 10 live B fragments and
    // staggered partial s_wait_loadcnt instead of full drains per k-step.
    __builtin_amdgcn_sched_barrier(0);

    // Dual-accumulator WMMA chain (breaks the serial RAW on the accumulator).
    v8f acc0 = {}, acc1 = {};
    #pragma unroll
    for (int t = 0; t < KSTEPS; t += 2) {
      acc0 = __builtin_amdgcn_wmma_f32_16x16x32_bf16(false, afr[t],     false, bfr[t],
                                                     (short)0, acc0, false, false);
      acc1 = __builtin_amdgcn_wmma_f32_16x16x32_bf16(false, afr[t + 1], false, bfr[t + 1],
                                                     (short)0, acc1, false, false);
    }
    v8f acc = acc0 + acc1;

    // stage 16x16 accumulator: C layout -> sh[Mrow*16 + Ncol]  (wave-private slice,
    // same-wave DS ops are in-order; compiler inserts the s_wait_dscnt for the reads)
    #pragma unroll
    for (int r = 0; r < 8; ++r) shw[(r + ksel * 8) * 16 + m] = acc[r];

    // lanes 0..15: each owns query row M=lane, scan 16 columns, maintain top-5
    if (lane < 16) {
      #pragma unroll
      for (int n = 0; n < 16; ++n) {
        float s = shw[lane * 16 + n];
        int   v = rowbase + n;
        if (v < VOCAB) insert5(tv, ti, s, v);
      }
    }
  }

  if (lane < 16) {
    size_t off = ((size_t)chunk * NQ + (qbase + lane)) * TOPK;
    #pragma unroll
    for (int j = 0; j < 5; ++j) { pvals[off + j] = tv[j]; pidx[off + j] = ti[j]; }
  }
}

// ---------------- K4: reduce per-chunk partials to final top-5 ----------------
__global__ __launch_bounds__(256) void final_topk_kernel(const float* __restrict__ pvals,
                                                         const int* __restrict__ pidx,
                                                         float* __restrict__ out) {
  __shared__ float sv[256 * 5];
  __shared__ int   si[256 * 5];
  const int q = blockIdx.x;
  const int t = threadIdx.x;

  float tv[5]; int ti[5];
  #pragma unroll
  for (int j = 0; j < 5; ++j) { tv[j] = -3.4e38f; ti[j] = 0; }

  for (int c = t; c < CHUNKS; c += 256) {
    size_t base = ((size_t)c * NQ + q) * TOPK;
    #pragma unroll
    for (int j = 0; j < 5; ++j) insert5(tv, ti, pvals[base + j], pidx[base + j]);
  }
  #pragma unroll
  for (int j = 0; j < 5; ++j) { sv[t * 5 + j] = tv[j]; si[t * 5 + j] = ti[j]; }
  __syncthreads();

  if (t < 32) {                         // 32-way merge: each merges 8 lists (40 entries)
    #pragma unroll
    for (int j = 0; j < 5; ++j) { tv[j] = -3.4e38f; ti[j] = 0; }
    for (int e = t * 40; e < t * 40 + 40; ++e) insert5(tv, ti, sv[e], si[e]);
  }
  __syncthreads();
  if (t < 32) {
    #pragma unroll
    for (int j = 0; j < 5; ++j) { sv[t * 5 + j] = tv[j]; si[t * 5 + j] = ti[j]; }
  }
  __syncthreads();

  if (t == 0) {                         // final merge of 160 entries
    #pragma unroll
    for (int j = 0; j < 5; ++j) { tv[j] = -3.4e38f; ti[j] = 0; }
    for (int e = 0; e < 160; ++e) insert5(tv, ti, sv[e], si[e]);
    #pragma unroll
    for (int j = 0; j < 5; ++j) {
      out[q * TOPK + j]             = tv[j];          // top values [512,5]
      out[NQ * TOPK + q * TOPK + j] = (float)ti[j];   // top indices [512,5] (exact in f32)
    }
  }
}

// ---------------- host launcher ----------------
extern "C" void kernel_launch(void* const* d_in, const int* in_sizes, int n_in,
                              void* d_out, int out_size, void* d_ws, size_t ws_size,
                              hipStream_t stream) {
  (void)in_sizes; (void)n_in; (void)out_size; (void)ws_size;
  const float* emb = (const float*)d_in[0];
  const int*   ids = (const int*)d_in[1];
  float*       out = (float*)d_out;

  // workspace layout (256B aligned slices)
  char* ws = (char*)d_ws;
  size_t o = 0;
  __bf16* ebf = (__bf16*)(ws + o); o += (size_t)VOCAB * KP * sizeof(__bf16);          // 256 MB
  o = (o + 255) & ~(size_t)255;
  __bf16* qbf = (__bf16*)(ws + o); o += (size_t)NQ * KP * sizeof(__bf16);             // 320 KB
  o = (o + 255) & ~(size_t)255;
  float* pvals = (float*)(ws + o); o += (size_t)CHUNKS * NQ * TOPK * sizeof(float);   // 16 MB
  o = (o + 255) & ~(size_t)255;
  int*   pidx  = (int*)(ws + o);   o += (size_t)CHUNKS * NQ * TOPK * sizeof(int);     // 16 MB

  norm_convert_kernel<<<(VOCAB + 7) / 8, 256, 0, stream>>>(emb, ebf);
  gather_queries_kernel<<<NQ, 64, 0, stream>>>(ebf, ids, qbf);
  dim3 grid2(QBLOCKS, CHUNKS);
  gemm_topk_kernel<<<grid2, 256, 0, stream>>>(ebf, qbf, pvals, pidx);
  final_topk_kernel<<<NQ, 256, 0, stream>>>(pvals, pidx, out);
}
